// RGCNEmbedding_20684562498295
// MI455X (gfx1250) — matrix operand order repaired and checked by reading, not compiled, not run on hardware
//
#include <hip/hip_runtime.h>
#include <cstdint>
#include <cstddef>

#define C_DIM   128
#define R_DIM   10
#define MSG_W   42   // 2*16 + 10
#define M_TILE  32   // node rows per block (two 16-row WMMA tiles)

typedef __attribute__((ext_vector_type(16))) __bf16 v16bf;
typedef __attribute__((ext_vector_type(8)))  float  v8f;

union ABFrag { v16bf v; uint4 q[2]; };

__device__ __forceinline__ unsigned short f2bf(float f) {
    unsigned int u = __builtin_bit_cast(unsigned int, f);
    u += 0x7FFFu + ((u >> 16) & 1u);          // round-to-nearest-even
    return (unsigned short)(u >> 16);
}

// ---------------------------------------------------------------- edge types
__global__ void rgcn_etype_kernel(const float* __restrict__ msg,
                                  int* __restrict__ etype, int E) {
    int e = blockIdx.x * blockDim.x + threadIdx.x;
    if (e >= E) return;
    const float* p = msg + (size_t)e * MSG_W + 16;
    float best = p[0]; int bi = 0;
#pragma unroll
    for (int j = 1; j < R_DIM; ++j) {
        float v = p[j];
        if (v > best) { best = v; bi = j; }   // first-max like jnp.argmax
    }
    etype[e] = bi;
}

// ------------------------------------------------- weights: f32 -> bf16, [slot][n][k]
__global__ void rgcn_wconv_kernel(const float* __restrict__ wrel,
                                  const float* __restrict__ root,
                                  unsigned short* __restrict__ wT) {
    int idx = blockIdx.x * blockDim.x + threadIdx.x;
    const int total = (R_DIM + 1) * C_DIM * C_DIM;
    if (idx >= total) return;
    int slot = idx >> 14;           // / (128*128)
    int rem  = idx & 16383;
    int o    = rem >> 7;            // output column (n)
    int k    = rem & 127;           // input dim (k)
    float v = (slot < R_DIM) ? wrel[(size_t)slot * C_DIM * C_DIM + (size_t)k * C_DIM + o]
                             : root[(size_t)k * C_DIM + o];
    wT[((size_t)slot * C_DIM + o) * C_DIM + k] = f2bf(v);
}

// ---------------------------------------------------------------- zero fill
__global__ void rgcn_fill4_kernel(float4* __restrict__ p, size_t n4) {
    size_t i = (size_t)blockIdx.x * blockDim.x + threadIdx.x;
    if (i < n4) p[i] = make_float4(0.f, 0.f, 0.f, 0.f);
}

// --------------------------------------------------- scatter: sums/counts per (rel,dst)
__global__ void rgcn_scatter_kernel(const float* __restrict__ xin,
                                    const int* __restrict__ src,
                                    const int* __restrict__ dst,
                                    const int* __restrict__ etype,
                                    float* __restrict__ sums,   // [nr][N][C]
                                    float* __restrict__ cnt,    // [nr][N]
                                    int N, int E, int r0, int nr) {
    long long t = (long long)blockIdx.x * blockDim.x + threadIdx.x;
    int e = (int)(t >> 5);
    if (e >= E) return;
    int lane = (int)(t & 31);
    int rr = etype[e] - r0;
    if (rr < 0 || rr >= nr) return;
    int s = src[e], d = dst[e];
    const float4 v = *(const float4*)(xin + (size_t)s * C_DIM + lane * 4);
    float* sb = sums + ((size_t)rr * N + d) * C_DIM + lane * 4;
    atomicAdd(sb + 0, v.x);
    atomicAdd(sb + 1, v.y);
    atomicAdd(sb + 2, v.z);
    atomicAdd(sb + 3, v.w);
    if (lane == 0) atomicAdd(cnt + (size_t)rr * N + d, 1.0f);
}

// --------------------------------------------------- WMMA GEMM + mean + bias + relu
// Block: 256 threads (8 waves). Block handles 32 node-rows (two 16-row WMMA tiles);
// wave w owns cols [16w,16w+16) with two f32 accumulators, reusing each B fragment
// for 2 WMMAs (weights stay L2-resident; A tiles staged in LDS as bf16).
__global__ __launch_bounds__(256)
void rgcn_gemm_kernel(const float* __restrict__ sums,   // [nr][N][C]
                      const float* __restrict__ cnt,    // [nr][N]
                      const float* __restrict__ xroot,  // [N][C] input for root term
                      const unsigned short* __restrict__ wT, // [11][C][C] bf16 (n-major)
                      const float* __restrict__ bias,
                      float* __restrict__ out,
                      int N, int r0, int nr,
                      int add_root, int first, int last, int do_relu) {
    __shared__ unsigned short aLds[M_TILE][C_DIM + 8];  // padded rows, 16B-aligned frags
    __shared__ float rowScale[M_TILE];

    const int tid  = threadIdx.x;
    const int lane = tid & 31;
    const int wave = tid >> 5;
    const int m0   = blockIdx.x * M_TILE;
    const int n0   = wave * 16;

    const int mA = lane & 15;                 // A row (within 16-row tile) this lane holds
    const int ka = (lane < 16) ? 0 : 8;       // A K-offset pattern (ISA 16-bit A table)
    const int kb = (lane < 16) ? 0 : 8;       // B K-offset pattern (same table order as A)
    const int nC = n0 + (lane & 15);          // global output column for B / C / D

    v8f acc0 = {};   // rows m0 +  0..15
    v8f acc1 = {};   // rows m0 + 16..31

    const int iters = nr + (add_root ? 1 : 0);
    for (int rel = 0; rel < iters; ++rel) {
        const bool isRoot = (rel == nr);
        const int  rr     = rel;                       // relative relation index
        const int  wslot  = isRoot ? R_DIM : (r0 + rel);

        __syncthreads();   // protect previous iteration's LDS reads
        if (tid < M_TILE) {
            float sc = 1.0f;
            if (!isRoot) {
                int gm = m0 + tid;
                float c = (gm < N) ? cnt[(size_t)rr * N + gm] : 1.0f;
                sc = __builtin_amdgcn_rcpf(fmaxf(c, 1.0f));   // v_rcp_f32
            }
            rowScale[tid] = sc;
        }
        __syncthreads();

        // stage 32x128 A tile (mean or x) into LDS as bf16
#pragma unroll
        for (int j = 0; j < (M_TILE * C_DIM) / 256; ++j) {
            int idx = tid + j * 256;         // 0..4095
            int m = idx >> 7, k = idx & 127;
            int gm = m0 + m;
            float v = 0.f;
            if (gm < N)
                v = isRoot ? xroot[(size_t)gm * C_DIM + k]
                           : sums[((size_t)rr * N + gm) * C_DIM + k];
            aLds[m][k] = f2bf(v * rowScale[m]);
        }
        __syncthreads();

        // 4 K-steps of 32; each B fragment feeds both row-tiles (2 WMMAs)
#pragma unroll
        for (int kk = 0; kk < 4; ++kk) {
            const int K0 = kk * 32;
            ABFrag a0, a1, b;
            a0.q[0] = *(const uint4*)&aLds[mA][K0 + ka];        // ds_read_b128
            a0.q[1] = *(const uint4*)&aLds[mA][K0 + 16 + ka];
            a1.q[0] = *(const uint4*)&aLds[16 + mA][K0 + ka];
            a1.q[1] = *(const uint4*)&aLds[16 + mA][K0 + 16 + ka];
            const unsigned short* bp =
                wT + ((size_t)wslot * C_DIM + nC) * C_DIM + K0 + kb;
            b.q[0] = ((const uint4*)bp)[0];                     // global_load_b128
            b.q[1] = ((const uint4*)bp)[2];
            acc0 = __builtin_amdgcn_wmma_f32_16x16x32_bf16(
                       false, a0.v, false, b.v, (short)0, acc0, false, false);
            acc1 = __builtin_amdgcn_wmma_f32_16x16x32_bf16(
                       false, a1.v, false, b.v, (short)0, acc1, false, false);
        }
    }

    // epilogue: accumulate across chunks, bias + relu on last chunk
    const int rowOff = (lane < 16) ? 0 : 8;
    const float bn = last ? bias[nC] : 0.f;
#pragma unroll
    for (int i = 0; i < 8; ++i) {
        int gm0 = m0 + rowOff + i;
        if (gm0 < N) {
            size_t o = (size_t)gm0 * C_DIM + nC;
            float v = acc0[i];
            if (!first) v += out[o];
            v += bn;
            if (last && do_relu) v = fmaxf(v, 0.f);
            out[o] = v;
        }
        int gm1 = m0 + 16 + rowOff + i;
        if (gm1 < N) {
            size_t o = (size_t)gm1 * C_DIM + nC;
            float v = acc1[i];
            if (!first) v += out[o];
            v += bn;
            if (last && do_relu) v = fmaxf(v, 0.f);
            out[o] = v;
        }
    }
}

// ----------------------------------------------------------------- launcher
extern "C" void kernel_launch(void* const* d_in, const int* in_sizes, int n_in,
                              void* d_out, int out_size, void* d_ws, size_t ws_size,
                              hipStream_t stream) {
    const float* x     = (const float*)d_in[0];
    // d_in[1] last_update, d_in[2] t : unused by the reference math
    const float* msg   = (const float*)d_in[3];
    const float* w1    = (const float*)d_in[4];
    const float* root1 = (const float*)d_in[5];
    const float* b1    = (const float*)d_in[6];
    const float* w2    = (const float*)d_in[7];
    const float* root2 = (const float*)d_in[8];
    const float* b2    = (const float*)d_in[9];
    const int*   eidx  = (const int*)d_in[10];

    const int N = in_sizes[1];          // last_update has N elements
    const int E = in_sizes[2];          // t has E elements
    const int* src = eidx;
    const int* dst = eidx + E;

    // workspace partition (256B aligned)
    char* wp = (char*)d_ws;
    auto alloc = [&](size_t bytes) -> void* {
        void* p = (void*)wp;
        wp += (bytes + 255) & ~(size_t)255;
        return p;
    };
    int*            etype = (int*)alloc((size_t)E * 4);
    float*          h     = (float*)alloc((size_t)N * C_DIM * 4);
    unsigned short* wbf1  = (unsigned short*)alloc((size_t)(R_DIM + 1) * C_DIM * C_DIM * 2);
    unsigned short* wbf2  = (unsigned short*)alloc((size_t)(R_DIM + 1) * C_DIM * C_DIM * 2);
    size_t used = (size_t)(wp - (char*)d_ws);

    // relation chunk size from remaining scratch: need nr*(N*C + N) floats
    size_t perRel = (size_t)N * C_DIM * 4 + (size_t)N * 4 + 512;
    size_t remain = (ws_size > used) ? (ws_size - used) : 0;
    int chunk = (int)(remain / perRel);
    if (chunk < 1)  chunk = 1;
    if (chunk > R_DIM) chunk = R_DIM;
    float* sums = (float*)alloc((size_t)chunk * N * C_DIM * 4);
    float* cnt  = (float*)alloc((size_t)chunk * N * 4);

    // 1) edge types
    rgcn_etype_kernel<<<(E + 255) / 256, 256, 0, stream>>>(msg, etype, E);

    // 2) bf16 transposed weights for both layers
    const int wTotal = (R_DIM + 1) * C_DIM * C_DIM;
    rgcn_wconv_kernel<<<(wTotal + 255) / 256, 256, 0, stream>>>(w1, root1, wbf1);
    rgcn_wconv_kernel<<<(wTotal + 255) / 256, 256, 0, stream>>>(w2, root2, wbf2);

    auto run_layer = [&](const float* xin, const unsigned short* wbf,
                         const float* bias, float* outp, int do_relu) {
        for (int r0 = 0; r0 < R_DIM; r0 += chunk) {
            const int nr = (R_DIM - r0 < chunk) ? (R_DIM - r0) : chunk;
            const size_t nSum4 = ((size_t)nr * N * C_DIM) / 4;
            const size_t nCnt4 = ((size_t)nr * N + 3) / 4;   // cnt alloc is 256B padded
            rgcn_fill4_kernel<<<(unsigned)((nSum4 + 255) / 256), 256, 0, stream>>>(
                (float4*)sums, nSum4);
            rgcn_fill4_kernel<<<(unsigned)((nCnt4 + 255) / 256), 256, 0, stream>>>(
                (float4*)cnt, nCnt4);
            const long long sThreads = (long long)E * 32;
            rgcn_scatter_kernel<<<(unsigned)((sThreads + 255) / 256), 256, 0, stream>>>(
                xin, src, dst, etype, sums, cnt, N, E, r0, nr);
            const int first = (r0 == 0);
            const int last  = (r0 + nr == R_DIM);
            rgcn_gemm_kernel<<<(N + M_TILE - 1) / M_TILE, 256, 0, stream>>>(
                sums, cnt, xin, wbf, bias, outp, N, r0, nr,
                /*add_root=*/first, first, last, do_relu);
        }
    };

    run_layer(x, wbf1, b1, h, /*relu=*/1);
    run_layer(h, wbf2, b2, (float*)d_out, /*relu=*/0);
}
